// TimeMix_40106404610566
// MI455X (gfx1250) — compile-verified, matrix operand order checked by reading, not run on hardware
//
#include <hip/hip_runtime.h>
#include <hip/hip_bf16.h>

#define Bv 4
#define Tv 1024
#define Cv 1024
#define Hv 16
#define Nv 64
#define DMIX 32
#define DDEC 64
#define BT  (Bv * Tv)          // 4096
#define BTC (Bv * Tv * Cv)     // 4194304

typedef __attribute__((ext_vector_type(16))) __bf16 v16bf;
typedef __attribute__((ext_vector_type(8)))  float  v8f;
typedef __attribute__((ext_vector_type(4))) unsigned int v4ui;
typedef __attribute__((ext_vector_type(8))) int v8si;
typedef __attribute__((ext_vector_type(4))) int v4si;

union BF16x16 { v16bf v; unsigned short u[16]; };

#if defined(__gfx1250__) && __has_builtin(__builtin_amdgcn_tensor_load_to_lds)
#define USE_TDM 1
#else
#define USE_TDM 0
#endif

__device__ __forceinline__ unsigned short f2bf(float f) {
  unsigned int x = __float_as_uint(f);
  unsigned int r = (x + 0x7FFFu + ((x >> 16) & 1u)) >> 16;   // RNE
  return (unsigned short)r;
}

#if USE_TDM
// 2D tile load via Tensor Data Mover: tile_d1 rows x tile_d0 elems of 2-byte data,
// row stride = stride_elems, packed contiguously into LDS at lds_off.
// D# layout per CDNA5 ISA (groups 0/1; groups 2/3 unused for 2D).
__device__ __forceinline__ void tdm_load_2d(unsigned lds_off, const void* gptr,
                                            unsigned tile_d0, unsigned tile_d1,
                                            unsigned tensor_d0, unsigned tensor_d1,
                                            unsigned stride_elems) {
  unsigned long long ga = (unsigned long long)gptr;
  v4ui g0;
  g0.x = 1u;                                    // count=1, user descriptor
  g0.y = lds_off;                               // D#.lds_addr
  g0.z = (unsigned)(ga & 0xFFFFFFFFu);          // D#.global_addr[31:0]
  g0.w = (unsigned)(ga >> 32) | 0x80000000u;    // global_addr[56:32] | type=2
  v8si g1;
  g1[0] = 0x10000;                              // data_size=1 (2 bytes); mask/flags=0
  g1[1] = (int)((tensor_d0 & 0xFFFFu) << 16);   // tensor_dim0[15:0]
  g1[2] = (int)((tensor_d0 >> 16) | ((tensor_d1 & 0xFFFFu) << 16));
  g1[3] = (int)((tensor_d1 >> 16) | (tile_d0 << 16));   // tensor_dim1[31:16]|tile_dim0
  g1[4] = (int)(tile_d1 & 0xFFFFu);             // tile_dim1 ; tile_dim2=0
  g1[5] = (int)stride_elems;                    // tensor_dim0_stride[31:0]
  g1[6] = 0;
  g1[7] = 0;
  v4si z4 = {0, 0, 0, 0};
  v8si z8 = {0, 0, 0, 0, 0, 0, 0, 0};
  __builtin_amdgcn_tensor_load_to_lds(g0, g1, z4, z4, z8, 0);   // 6-arg form
}
#endif

// ---------------------------------------------------------------- prep
__global__ void prep_kernel(const float* __restrict__ x, const float* __restrict__ shift,
                            const float* __restrict__ maa_x,
                            float* __restrict__ dxprev, float* __restrict__ xxx) {
  int i = blockIdx.x * blockDim.x + threadIdx.x;
  if (i >= BTC) return;
  int c  = i & (Cv - 1);
  int bt = i >> 10;
  int t  = bt & (Tv - 1);
  int b  = bt >> 10;
  float xi   = x[i];
  float prev = (t == 0) ? shift[b * Cv + c] : x[i - Cv];
  float dx   = prev - xi;
  dxprev[i]  = dx;
  xxx[i]     = xi + dx * maa_x[c];
}

// ---------------------------------------------------------------- weight transpose + f32->bf16 (once per weight)
__global__ void wtrans_kernel(const float* __restrict__ W, unsigned short* __restrict__ Wt) {
  __shared__ float tile[32][33];
  int k0 = blockIdx.x * 32, n0 = blockIdx.y * 32;
  int tx = threadIdx.x & 31, ty = threadIdx.x >> 5;        // 256 thr: ty 0..7
  #pragma unroll
  for (int i = ty; i < 32; i += 8)
    tile[i][tx] = W[(size_t)(k0 + i) * Cv + n0 + tx];
  __syncthreads();
  #pragma unroll
  for (int i = ty; i < 32; i += 8)
    Wt[(size_t)(n0 + i) * Cv + k0 + tx] = f2bf(tile[tx][i]);  // Wt[n][k] = W[k][n]
}

// ---------------------------------------------------------------- lora1: tmp5 = tanh(xxx @ w1)  [BT,160]
__global__ void lora1_kernel(const float* __restrict__ xxx, const float* __restrict__ w1,
                             float* __restrict__ tmp5) {
  __shared__ float sx[Cv];
  int bt = blockIdx.x;
  const float* xp = xxx + (size_t)bt * Cv;
  for (int k = threadIdx.x; k < Cv; k += blockDim.x) sx[k] = xp[k];
  __syncthreads();
  int o = threadIdx.x;                 // 0..159
  float acc = 0.f;
  #pragma unroll 8
  for (int k = 0; k < Cv; ++k) acc = fmaf(sx[k], w1[(size_t)k * 160 + o], acc);
  tmp5[(size_t)bt * 160 + o] = tanhf(acc);
}

// ---------------------------------------------------------------- mix: five mixed inputs
__global__ void mix_kernel(const float* __restrict__ x, const float* __restrict__ dxprev,
                           const float* __restrict__ tmp5, const float* __restrict__ w2,
                           const float* __restrict__ maa_w, const float* __restrict__ maa_k,
                           const float* __restrict__ maa_v, const float* __restrict__ maa_r,
                           const float* __restrict__ maa_g,
                           float* __restrict__ xw,
                           unsigned short* __restrict__ xk, unsigned short* __restrict__ xv,
                           unsigned short* __restrict__ xr, unsigned short* __restrict__ xg) {
  __shared__ float st[5 * DMIX];
  int bt = blockIdx.y;
  int c  = blockIdx.x * blockDim.x + threadIdx.x;
  if (threadIdx.x < 5 * DMIX) st[threadIdx.x] = tmp5[(size_t)bt * 160 + threadIdx.x];
  __syncthreads();
  size_t i = (size_t)bt * Cv + c;
  float m[5];
  #pragma unroll
  for (int s = 0; s < 5; ++s) {
    float a = 0.f;
    #pragma unroll
    for (int d = 0; d < DMIX; ++d)
      a = fmaf(st[s * DMIX + d], w2[(size_t)(s * DMIX + d) * Cv + c], a);
    m[s] = a;
  }
  float xi = x[i], dx = dxprev[i];
  xw[i] = xi + dx * (maa_w[c] + m[0]);
  xk[i] = f2bf(xi + dx * (maa_k[c] + m[1]));
  xv[i] = f2bf(xi + dx * (maa_v[c] + m[2]));
  xr[i] = f2bf(xi + dx * (maa_r[c] + m[3]));
  xg[i] = f2bf(xi + dx * (maa_g[c] + m[4]));
}

// ---------------------------------------------------------------- lora2: t2 = tanh(xw @ dw1)  [BT,64]
__global__ void lora2_kernel(const float* __restrict__ xw, const float* __restrict__ dw1,
                             float* __restrict__ t2) {
  __shared__ float sx[Cv];
  int bt = blockIdx.x;
  const float* xp = xw + (size_t)bt * Cv;
  for (int k = threadIdx.x; k < Cv; k += blockDim.x) sx[k] = xp[k];
  __syncthreads();
  int o = threadIdx.x;                 // 0..63
  float acc = 0.f;
  #pragma unroll 8
  for (int k = 0; k < Cv; ++k) acc = fmaf(sx[k], dw1[(size_t)k * DDEC + o], acc);
  t2[(size_t)bt * DDEC + o] = tanhf(acc);
}

// ---------------------------------------------------------------- decay = exp(-exp(td + t2 @ dw2))
__global__ void decay_kernel(const float* __restrict__ t2, const float* __restrict__ dw2,
                             const float* __restrict__ td, float* __restrict__ decay) {
  __shared__ float st[DDEC];
  int bt = blockIdx.y;
  int c  = blockIdx.x * blockDim.x + threadIdx.x;
  if (threadIdx.x < DDEC) st[threadIdx.x] = t2[(size_t)bt * DDEC + threadIdx.x];
  __syncthreads();
  float w = td[c];
  #pragma unroll
  for (int d = 0; d < DDEC; ++d) w = fmaf(st[d], dw2[(size_t)d * Cv + c], w);
  decay[(size_t)bt * Cv + c] = expf(-expf(w));
}

// ---------------------------------------------------------------- WMMA GEMM: Y[M,N] = A(bf16 [M,K]) @ Wt(bf16 [N,K])^T
// 256 threads = 8 waves; block tile 64(M) x 128(N); wave tile 16x64 = 4 WMMA accumulators.
// Tiles staged into LDS by the Tensor Data Mover (wave 0), compute overlap via workgroup barrier.
__global__ __launch_bounds__(256)
void gemm_bf16_kernel(const unsigned short* __restrict__ A, const unsigned short* __restrict__ Wt,
                      float* __restrict__ Y, int M, int N, int K, int mode) {
  __shared__ __align__(16) unsigned short As[64][32];
  __shared__ __align__(16) unsigned short Bs[128][32];   // Bs[n][k]
  int tid  = threadIdx.x;
  int lane = tid & 31;
  int w    = tid >> 5;
  int wm   = w >> 1, wn = w & 1;                // wave tile: rows 16*wm, cols 64*wn
  int m0 = blockIdx.y * 64;
  int n0 = blockIdx.x * 128;
  int l  = lane & 15;
  int kh = (lane >> 4) << 3;                    // 0 or 8 (ISA 16-bit A/B lane layout)
  v8f acc[4] = {v8f{}, v8f{}, v8f{}, v8f{}};
#if USE_TDM
  unsigned as_off = (unsigned)(unsigned long long)(void*)&As[0][0];
  unsigned bs_off = (unsigned)(unsigned long long)(void*)&Bs[0][0];
#else
  int arow = tid >> 2, acol = (tid & 3) << 3;   // A: 8 contiguous bf16 per thread
  int brow = tid >> 1, bcol = (tid & 1) << 4;   // B: 16 contiguous bf16 per thread
#endif
  for (int kk = 0; kk < K; kk += 32) {
#if USE_TDM
    if (w == 0) {                                // one wave drives the TDM
      tdm_load_2d(as_off, A  + (size_t)m0 * K + kk, 32, 64,  (unsigned)K, (unsigned)M, (unsigned)K);
      tdm_load_2d(bs_off, Wt + (size_t)n0 * K + kk, 32, 128, (unsigned)K, (unsigned)N, (unsigned)K);
      __builtin_amdgcn_s_wait_tensorcnt(0);
    }
    __syncthreads();
#else
    *reinterpret_cast<uint4*>(&As[arow][acol]) =
        *reinterpret_cast<const uint4*>(A + (size_t)(m0 + arow) * K + kk + acol);
    *reinterpret_cast<uint4*>(&Bs[brow][bcol]) =
        *reinterpret_cast<const uint4*>(Wt + (size_t)(n0 + brow) * K + kk + bcol);
    *reinterpret_cast<uint4*>(&Bs[brow][bcol + 8]) =
        *reinterpret_cast<const uint4*>(Wt + (size_t)(n0 + brow) * K + kk + bcol + 8);
    if (kk + 32 < K)
      __builtin_prefetch(A + (size_t)(m0 + arow) * K + kk + 32 + acol, 0, 1);
    __syncthreads();
#endif
    BF16x16 af;
    const unsigned short* ap = &As[(wm << 4) + l][0];
    #pragma unroll
    for (int i = 0; i < 8; ++i) {
      af.u[i]     = ap[kh + i];
      af.u[8 + i] = ap[16 + kh + i];
    }
    #pragma unroll
    for (int s = 0; s < 4; ++s) {
      BF16x16 bf;
      const unsigned short* bp = &Bs[(wn << 6) + (s << 4) + l][0];
      #pragma unroll
      for (int i = 0; i < 8; ++i) {
        bf.u[i]     = bp[kh + i];
        bf.u[8 + i] = bp[16 + kh + i];
      }
      acc[s] = __builtin_amdgcn_wmma_f32_16x16x32_bf16(false, af.v, false, bf.v,
                                                       (short)0, acc[s], false, false);
    }
    __syncthreads();
  }
  int nOut = lane & 15;
  int mb   = (wm << 4) + ((lane >> 4) << 3);    // D layout: VGPR r -> row r (+8 for lanes 16-31)
  #pragma unroll
  for (int s = 0; s < 4; ++s) {
    int ncol = n0 + (wn << 6) + (s << 4) + nOut;
    #pragma unroll
    for (int r = 0; r < 8; ++r) {
      float v0 = acc[s][r];
      if (mode == 1) v0 = v0 / (1.f + expf(-v0));   // fused silu (g projection)
      Y[(size_t)(m0 + mb + r) * N + ncol] = v0;
    }
  }
}

// ---------------------------------------------------------------- WKV6 scan: one block per (b,h), thread j owns column j
__global__ __launch_bounds__(64)
void wkv_scan_kernel(const float* __restrict__ r, const float* __restrict__ k,
                     const float* __restrict__ v, const float* __restrict__ dec,
                     const float* __restrict__ faaaa, const float* __restrict__ state0,
                     float* __restrict__ y) {
  int bh = blockIdx.x;
  int b = bh >> 4, h = bh & 15;
  int j = threadIdx.x;
  float S[Nv];
  const float* s0 = state0 + (size_t)h * Nv * Nv;
  #pragma unroll
  for (int i = 0; i < Nv; ++i) S[i] = s0[i * Nv + j];
  __shared__ float sr[Nv], sk[Nv], sd[Nv], su[Nv];
  su[j] = faaaa[h * Nv + j];
  __syncthreads();
  size_t base = (size_t)b * Tv * Cv + h * Nv + j;
  for (int t = 0; t < Tv; ++t) {
    size_t idx = base + (size_t)t * Cv;
    sr[j] = r[idx];
    sk[j] = k[idx];
    sd[j] = dec[idx];
    float vj = v[idx];
    __syncthreads();
    float yj = 0.f;
    #pragma unroll
    for (int i = 0; i < Nv; ++i) {       // fully unrolled -> S stays in VGPRs
      float kv = sk[i] * vj;
      yj   = fmaf(sr[i], fmaf(su[i], kv, S[i]), yj);
      S[i] = fmaf(sd[i], S[i], kv);
    }
    y[idx] = yj;
    __syncthreads();
  }
}

// ---------------------------------------------------------------- GroupNorm(H groups) * g -> bf16
__global__ void gn_kernel(const float* __restrict__ y, const float* __restrict__ g,
                          const float* __restrict__ lw, const float* __restrict__ lb,
                          unsigned short* __restrict__ yg) {
  int gid = blockIdx.x * blockDim.x + threadIdx.x;   // (bt, h)
  if (gid >= BT * Hv) return;
  int h = gid & 15;
  int bt = gid >> 4;
  const float* yp = y + (size_t)bt * Cv + h * Nv;
  float mu = 0.f;
  #pragma unroll 8
  for (int j = 0; j < Nv; ++j) mu += yp[j];
  mu *= (1.f / Nv);
  float var = 0.f;
  #pragma unroll 8
  for (int j = 0; j < Nv; ++j) { float d = yp[j] - mu; var = fmaf(d, d, var); }
  var *= (1.f / Nv);
  float inv = rsqrtf(var + 6.4e-4f);                 // eps = 1e-5 * 8^2
  const float* gp = g + (size_t)bt * Cv + h * Nv;
  unsigned short* op = yg + (size_t)bt * Cv + h * Nv;
  #pragma unroll 8
  for (int j = 0; j < Nv; ++j) {
    float yn = (yp[j] - mu) * inv * lw[h * Nv + j] + lb[h * Nv + j];
    op[j] = f2bf(yn * gp[j]);
  }
}

// ================================================================ host
extern "C" void kernel_launch(void* const* d_in, const int* in_sizes, int n_in,
                              void* d_out, int out_size, void* d_ws, size_t ws_size,
                              hipStream_t stream) {
  const float* x      = (const float*)d_in[0];
  const float* shift  = (const float*)d_in[1];
  const float* maa_x  = (const float*)d_in[2];
  const float* maa_w  = (const float*)d_in[3];
  const float* maa_k  = (const float*)d_in[4];
  const float* maa_v  = (const float*)d_in[5];
  const float* maa_r  = (const float*)d_in[6];
  const float* maa_g  = (const float*)d_in[7];
  const float* w1     = (const float*)d_in[8];
  const float* w2     = (const float*)d_in[9];
  const float* td     = (const float*)d_in[10];
  const float* dw1    = (const float*)d_in[11];
  const float* dw2    = (const float*)d_in[12];
  const float* faaaa  = (const float*)d_in[13];
  const float* state0 = (const float*)d_in[14];
  const float* W_r    = (const float*)d_in[15];
  const float* W_k    = (const float*)d_in[16];
  const float* W_v    = (const float*)d_in[17];
  const float* W_g    = (const float*)d_in[18];
  const float* W_o    = (const float*)d_in[19];
  const float* ln_w   = (const float*)d_in[20];
  const float* ln_b   = (const float*)d_in[21];
  float* out = (float*)d_out;

  char* p = (char*)d_ws;
  size_t off = 0;
  auto take = [&](size_t bytes) { size_t o = off; off += (bytes + 255) & ~(size_t)255; return o; };
  size_t o_dxprev = take((size_t)BTC * 4);          // reused as `decay` after mix
  size_t o_xxx    = take((size_t)BTC * 4);          // reused as `y` after lora1
  size_t o_tmp5   = take((size_t)BT * 160 * 4);
  size_t o_xw     = take((size_t)BTC * 4);
  size_t o_t2     = take((size_t)BT * DDEC * 4);
  size_t o_xr     = take((size_t)BTC * 2);
  size_t o_xk     = take((size_t)BTC * 2);
  size_t o_xv     = take((size_t)BTC * 2);
  size_t o_xg     = take((size_t)BTC * 2);
  size_t o_r      = take((size_t)BTC * 4);
  size_t o_k      = take((size_t)BTC * 4);
  size_t o_v      = take((size_t)BTC * 4);
  size_t o_g      = take((size_t)BTC * 4);
  size_t o_yg     = take((size_t)BTC * 2);
  size_t o_wtr    = take((size_t)Cv * Cv * 2);
  size_t o_wtk    = take((size_t)Cv * Cv * 2);
  size_t o_wtv    = take((size_t)Cv * Cv * 2);
  size_t o_wtg    = take((size_t)Cv * Cv * 2);
  size_t o_wto    = take((size_t)Cv * Cv * 2);

  float* dxprev = (float*)(p + o_dxprev);
  float* xxx    = (float*)(p + o_xxx);
  float* tmp5   = (float*)(p + o_tmp5);
  float* xw     = (float*)(p + o_xw);
  float* t2     = (float*)(p + o_t2);
  unsigned short* xr = (unsigned short*)(p + o_xr);
  unsigned short* xk = (unsigned short*)(p + o_xk);
  unsigned short* xv = (unsigned short*)(p + o_xv);
  unsigned short* xg = (unsigned short*)(p + o_xg);
  float* rb = (float*)(p + o_r);
  float* kb = (float*)(p + o_k);
  float* vb = (float*)(p + o_v);
  float* gb = (float*)(p + o_g);
  unsigned short* yg   = (unsigned short*)(p + o_yg);
  unsigned short* wt_r = (unsigned short*)(p + o_wtr);
  unsigned short* wt_k = (unsigned short*)(p + o_wtk);
  unsigned short* wt_v = (unsigned short*)(p + o_wtv);
  unsigned short* wt_g = (unsigned short*)(p + o_wtg);
  unsigned short* wt_o = (unsigned short*)(p + o_wto);
  float* decay = dxprev;   // overlay: dxprev dead after mix_kernel
  float* ybuf  = xxx;      // overlay: xxx dead after lora1_kernel

  dim3 gt(Cv / 32, Cv / 32);
  wtrans_kernel<<<gt, 256, 0, stream>>>(W_r, wt_r);
  wtrans_kernel<<<gt, 256, 0, stream>>>(W_k, wt_k);
  wtrans_kernel<<<gt, 256, 0, stream>>>(W_v, wt_v);
  wtrans_kernel<<<gt, 256, 0, stream>>>(W_g, wt_g);
  wtrans_kernel<<<gt, 256, 0, stream>>>(W_o, wt_o);

  prep_kernel<<<BTC / 256, 256, 0, stream>>>(x, shift, maa_x, dxprev, xxx);
  lora1_kernel<<<BT, 160, 0, stream>>>(xxx, w1, tmp5);
  dim3 gmix(Cv / 256, BT);
  mix_kernel<<<gmix, 256, 0, stream>>>(x, dxprev, tmp5, w2, maa_w, maa_k, maa_v, maa_r, maa_g,
                                       xw, xk, xv, xr, xg);
  lora2_kernel<<<BT, 64, 0, stream>>>(xw, dw1, t2);
  decay_kernel<<<gmix, 256, 0, stream>>>(t2, dw2, td, decay);

  dim3 gg(Cv / 128, BT / 64);           // (8, 64)
  gemm_bf16_kernel<<<gg, 256, 0, stream>>>(xr, wt_r, rb, BT, Cv, Cv, 0);
  gemm_bf16_kernel<<<gg, 256, 0, stream>>>(xk, wt_k, kb, BT, Cv, Cv, 0);
  gemm_bf16_kernel<<<gg, 256, 0, stream>>>(xv, wt_v, vb, BT, Cv, Cv, 0);
  gemm_bf16_kernel<<<gg, 256, 0, stream>>>(xg, wt_g, gb, BT, Cv, Cv, 1);  // fused silu

  wkv_scan_kernel<<<Bv * Hv, Nv, 0, stream>>>(rb, kb, vb, decay, faaaa, state0, ybuf);
  gn_kernel<<<(BT * Hv + 255) / 256, 256, 0, stream>>>(ybuf, gb, ln_w, ln_b, yg);
  gemm_bf16_kernel<<<gg, 256, 0, stream>>>(yg, wt_o, out, BT, Cv, Cv, 0);
}
